// Fp8MLP_72541997629815
// MI455X (gfx1250) — 8-bit matrix operand order repaired and checked by reading, compiled, not run on hardware
//
#include <hip/hip_runtime.h>

typedef __attribute__((ext_vector_type(16))) int   v16i;
typedef __attribute__((ext_vector_type(8)))  float v8f;

#define FP8_MAX_F 57344.0f

#if __has_builtin(__builtin_amdgcn_wmma_f32_16x16x128_bf8_bf8)
#define WMMA_E5M2(a, b, c) __builtin_amdgcn_wmma_f32_16x16x128_bf8_bf8((a), (b), (short)0, (c), false, false)
#else
#define WMMA_E5M2(a, b, c) __builtin_amdgcn_wmma_f32_16x16x128_fp8_fp8((a), (b), (short)0, (c), false, false)
#endif

__device__ __forceinline__ unsigned int f32_to_e5m2(float v) {
    _Float16 h = (_Float16)v;
    unsigned short b = __builtin_bit_cast(unsigned short, h);
    unsigned short r = (unsigned short)(b + 0x7Fu + ((b >> 8) & 1u));
    return (unsigned int)(r >> 8) & 0xFFu;
}


__global__ void __launch_bounds__(256) fp8mlp_quant_act(
    const float* __restrict__ x, unsigned char* __restrict__ q,
    float* __restrict__ s, int Kb, int M)
{
    const int lane = threadIdx.x & 31;
    const int wid  = blockIdx.x * (blockDim.x >> 5) + (threadIdx.x >> 5);
    const int row  = wid / Kb;
    const int kb   = wid - row * Kb;
    const size_t K = (size_t)Kb * 128;

    const float4 v = *(const float4*)(x + (size_t)row * K + kb * 128 + lane * 4);
    float m = fmaxf(fmaxf(fabsf(v.x), fabsf(v.y)), fmaxf(fabsf(v.z), fabsf(v.w)));
#pragma unroll
    for (int off = 16; off >= 1; off >>= 1)
        m = fmaxf(m, __shfl_xor(m, off, 32));

    const float sc  = m * (1.0f / FP8_MAX_F);
    const float inv = (sc > 0.0f) ? (1.0f / sc) : 0.0f;
    unsigned int pk = f32_to_e5m2(v.x * inv)
                    | (f32_to_e5m2(v.y * inv) << 8)
                    | (f32_to_e5m2(v.z * inv) << 16)
                    | (f32_to_e5m2(v.w * inv) << 24);
    *(unsigned int*)(q + (size_t)row * K + kb * 128 + lane * 4) = pk;
    if (lane == 0) s[(size_t)kb * M + row] = sc;
}

__global__ void __launch_bounds__(256) fp8mlp_quant_weight(
    const float* __restrict__ w, unsigned char* __restrict__ q,
    float* __restrict__ s, int K)
{
    const int kb = blockIdx.x, nb = blockIdx.y, Kb = gridDim.x;
    const int t  = threadIdx.x;
    const int row = nb * 128 + (t >> 1);
    const int col = kb * 128 + (t & 1) * 64;

    const float4* p = (const float4*)(w + (size_t)row * K + col);
    float4 vals[16];
    float m = 0.0f;
#pragma unroll
    for (int i = 0; i < 16; ++i) {
        vals[i] = p[i];
        m = fmaxf(m, fmaxf(fmaxf(fabsf(vals[i].x), fabsf(vals[i].y)),
                           fmaxf(fabsf(vals[i].z), fabsf(vals[i].w))));
    }
    __shared__ float red[256];
    red[t] = m;
    __syncthreads();
#pragma unroll
    for (int st = 128; st >= 1; st >>= 1) {
        if (t < st) red[t] = fmaxf(red[t], red[t + st]);
        __syncthreads();
    }
    const float sc  = red[0] * (1.0f / FP8_MAX_F);
    const float inv = (sc > 0.0f) ? (1.0f / sc) : 0.0f;

    unsigned int* qp = (unsigned int*)(q + (size_t)row * K + col);
#pragma unroll
    for (int i = 0; i < 16; ++i) {
        qp[i] = f32_to_e5m2(vals[i].x * inv)
              | (f32_to_e5m2(vals[i].y * inv) << 8)
              | (f32_to_e5m2(vals[i].z * inv) << 16)
              | (f32_to_e5m2(vals[i].w * inv) << 24);
    }
    if (t == 0) s[(size_t)nb * Kb + kb] = sc;
}


__device__ __forceinline__ v16i load_a16x128(const unsigned char* p) {
    union { v16i v; uint4 d[4]; } u;
#pragma unroll
    for (int c = 0; c < 4; ++c) u.d[c] = *(const uint4*)(p + c * 32);
    return u.v;
}

#define BSTRIDE 144
#define BSTAGE  (128 * BSTRIDE)

__device__ __forceinline__ void fill_b(unsigned char* dst, const unsigned char* __restrict__ Bq,
                                       int K, int nrow0, int kbyte, int t) {
    const uint4* src = (const uint4*)(Bq + (size_t)(nrow0 + (t >> 1)) * K + kbyte + (t & 1) * 64);
    uint4* d = (uint4*)(dst + (t >> 1) * BSTRIDE + (t & 1) * 64);
#pragma unroll
    for (int i = 0; i < 4; ++i) d[i] = src[i];
}

__device__ __forceinline__ void gemm_acc_lds(
    v8f acc[8], unsigned char (*bbuf)[BSTAGE],
    const unsigned char* __restrict__ Aq, const float* __restrict__ AsT,
    const unsigned char* __restrict__ Bq, const float* __restrict__ Bs,
    int K, int Kb, int M, int mw, int nrow0, int nScaleBlk,
    int half, int l15, int t)
{
    const unsigned char* aRow = Aq + (size_t)(mw + l15) * K + half * 16;
    fill_b(bbuf[0], Bq, K, nrow0, 0, t);
    for (int kb = 0; kb < Kb; ++kb) {
        const int kbyte = kb * 128;
        __syncthreads();
        if (kb + 1 < Kb) fill_b(bbuf[(kb + 1) & 1], Bq, K, nrow0, kbyte + 128, t);
        if (kb + 1 < Kb) __builtin_prefetch(aRow + kbyte + 128, 0, 0);

        v16i a = load_a16x128(aRow + kbyte);

        const float sB = Bs[(size_t)nScaleBlk * Kb + kb];
        const float* sp = AsT + (size_t)kb * M + mw + half * 8;
        const float4 s0 = *(const float4*)(sp);
        const float4 s1 = *(const float4*)(sp + 4);
        const float sab[8] = { s0.x * sB, s0.y * sB, s0.z * sB, s0.w * sB,
                               s1.x * sB, s1.y * sB, s1.z * sB, s1.w * sB };

        const unsigned char* bl = bbuf[kb & 1];
#pragma unroll
        for (int j = 0; j < 8; ++j) {
            const unsigned char* bp = bl + (16 * j + l15) * BSTRIDE + half * 16;
            union { v16i v; uint4 d[4]; } u;
#pragma unroll
            for (int g = 0; g < 4; ++g) u.d[g] = *(const uint4*)(bp + g * 32);
            v8f z = {};
            v8f p = WMMA_E5M2(a, u.v, z);
#pragma unroll
            for (int v = 0; v < 8; ++v)
                acc[j][v] = fmaf(sab[v], p[v], acc[j][v]);
        }
    }
    __syncthreads();
}

__global__ void __launch_bounds__(256) fp8mlp_gemm1_swiglu_quant(
    const unsigned char* __restrict__ xq, const float* __restrict__ xsT,
    const unsigned char* __restrict__ wq, const float* __restrict__ wsc,
    unsigned char* __restrict__ actq, float* __restrict__ actsT)
{
    const int K = 4096, Kb = 32, I = 11008, M = 4096;
    __shared__ float gateLds[128 * 128];
    __shared__ unsigned char bbuf[2][BSTAGE];

    const int t = threadIdx.x;
    const int lane = t & 31, wave = t >> 5;
    const int half = lane >> 4, l15 = lane & 15;
    const int n0 = blockIdx.x * 128;
    const int mw = blockIdx.y * 128 + wave * 16;

    v8f acc[8] = {};
    gemm_acc_lds(acc, bbuf, xq, xsT, wq, wsc, K, Kb, M,
                 mw, I + n0, (I + n0) >> 7, half, l15, t);
    const int rbase = wave * 16 + half * 8;
#pragma unroll
    for (int j = 0; j < 8; ++j)
#pragma unroll
        for (int v = 0; v < 8; ++v)
            gateLds[(rbase + v) * 128 + 16 * j + l15] = acc[j][v];

    v8f z = {};
#pragma unroll
    for (int j = 0; j < 8; ++j) acc[j] = z;
    gemm_acc_lds(acc, bbuf, xq, xsT, wq, wsc, K, Kb, M,
                 mw, n0, n0 >> 7, half, l15, t);

#pragma unroll
    for (int j = 0; j < 8; ++j)
#pragma unroll
        for (int v = 0; v < 8; ++v) {
            float g = gateLds[(rbase + v) * 128 + 16 * j + l15];
            acc[j][v] = acc[j][v] * g / (1.0f + __expf(-g));
        }

    float inv[8];
    const int nblk = n0 >> 7;
#pragma unroll
    for (int v = 0; v < 8; ++v) {
        float m = 0.0f;
#pragma unroll
        for (int j = 0; j < 8; ++j) m = fmaxf(m, fabsf(acc[j][v]));
#pragma unroll
        for (int off = 8; off >= 1; off >>= 1)
            m = fmaxf(m, __shfl_xor(m, off, 32));
        const float sc = m * (1.0f / FP8_MAX_F);
        inv[v] = (sc > 0.0f) ? (1.0f / sc) : 0.0f;
        if (l15 == 0)
            actsT[(size_t)nblk * M + (mw + half * 8 + v)] = sc;
    }
#pragma unroll
    for (int j = 0; j < 8; ++j)
#pragma unroll
        for (int v = 0; v < 8; ++v)
            actq[(size_t)(mw + half * 8 + v) * I + n0 + 16 * j + l15] =
                (unsigned char)f32_to_e5m2(acc[j][v] * inv[v]);
}

__global__ void __launch_bounds__(256) fp8mlp_gemm2(
    const unsigned char* __restrict__ aq, const float* __restrict__ asT,
    const unsigned char* __restrict__ wq, const float* __restrict__ wsc,
    float* __restrict__ out)
{
    const int K = 11008, Kb = 86, N = 4096, M = 4096;
    __shared__ unsigned char bbuf[2][BSTAGE];

    const int t = threadIdx.x;
    const int lane = t & 31, wave = t >> 5;
    const int half = lane >> 4, l15 = lane & 15;
    const int n0 = blockIdx.x * 128;
    const int mw = blockIdx.y * 128 + wave * 16;

    v8f acc[8] = {};
    gemm_acc_lds(acc, bbuf, aq, asT, wq, wsc, K, Kb, M,
                 mw, n0, n0 >> 7, half, l15, t);

#pragma unroll
    for (int j = 0; j < 8; ++j)
#pragma unroll
        for (int v = 0; v < 8; ++v)
            out[(size_t)(mw + half * 8 + v) * N + n0 + 16 * j + l15] = acc[j][v];
}


extern "C" void kernel_launch(void* const* d_in, const int* in_sizes, int n_in,
                              void* d_out, int out_size, void* d_ws, size_t ws_size,
                              hipStream_t stream) {
    (void)in_sizes; (void)n_in; (void)out_size; (void)ws_size;
    const float* x        = (const float*)d_in[0];
    const float* w_upgate = (const float*)d_in[1];
    const float* w_down   = (const float*)d_in[2];
    float* out = (float*)d_out;

    const size_t M = 4096, H = 4096, I = 11008, N1 = 22016;
    auto alignUp = [](size_t v) { return (v + 255) & ~(size_t)255; };
    unsigned char* ws = (unsigned char*)d_ws;
    size_t off = 0;
    unsigned char* xq  = ws + off;           off = alignUp(off + M * H);
    float*         xsT = (float*)(ws + off); off = alignUp(off + 32 * M * 4);
    unsigned char* wq1 = ws + off;           off = alignUp(off + N1 * H);
    float*         ws1 = (float*)(ws + off); off = alignUp(off + 172 * 32 * 4);
    unsigned char* wq2 = ws + off;           off = alignUp(off + H * I);
    float*         ws2 = (float*)(ws + off); off = alignUp(off + 32 * 86 * 4);
    unsigned char* aq  = ws + off;           off = alignUp(off + M * I);
    float*         asT = (float*)(ws + off); off = alignUp(off + 86 * M * 4);

    fp8mlp_quant_act<<<dim3((unsigned)(M * 32 / 8)), 256, 0, stream>>>(x, xq, xsT, 32, (int)M);
    fp8mlp_quant_weight<<<dim3(32, 172), 256, 0, stream>>>(w_upgate, wq1, ws1, 4096);
    fp8mlp_quant_weight<<<dim3(86, 32),  256, 0, stream>>>(w_down,   wq2, ws2, 11008);
    fp8mlp_gemm1_swiglu_quant<<<dim3(86, 32), 256, 0, stream>>>(xq, xsT, wq1, ws1, aq, asT);
    fp8mlp_gemm2<<<dim3(32, 32), 256, 0, stream>>>(aq, asT, wq2, ws2, out);
}
